// GNNEncoderB_26113401160169
// MI455X (gfx1250) — matrix operand order repaired and checked by reading, not compiled, not run on hardware
//
#include <hip/hip_runtime.h>

#define NB 32
#define CH 64

typedef __attribute__((ext_vector_type(16))) __bf16 v16bf;
typedef __attribute__((ext_vector_type(8)))  float  v8f;
typedef __attribute__((ext_vector_type(4)))  unsigned int u32x4;
typedef __attribute__((ext_vector_type(8)))  int i32x8;
typedef __attribute__((ext_vector_type(4)))  int i32x4;

#if __has_builtin(__builtin_amdgcn_tensor_load_to_lds) && __has_builtin(__builtin_amdgcn_s_wait_tensorcnt)
#define USE_TDM 1
#else
#define USE_TDM 0
#endif

// ---- WMMA fragment loaders (contiguous k = 16h + s order, same in A and B) -------

// A 16x32 bf16: lanes 0-15 rows M=0..15; VGPR v: K=2v,2v+1 (+16 for hi lanes).
__device__ inline v16bf load_a_frag(const __bf16* row, int kt, int hi) {
  v16bf a;
#pragma unroll
  for (int v = 0; v < 8; ++v) {
    int kb = kt * 32 + (hi << 4) + 2 * v;
    a[2 * v]     = row[kb];
    a[2 * v + 1] = row[kb + 1];
  }
  return a;
}

// B 32x16 bf16: lanes 0-15 cols N, K=0..15 across 8 VGPRs (2/VGPR); lanes 16-31 K=16..31.
__device__ inline v16bf load_b_frag(const __bf16* W, int kt, int nt, int lane) {
  int ncol = nt * 16 + (lane & 15);
  int koff = kt * 32 + ((lane >> 4) << 4);
  v16bf bm;
#pragma unroll
  for (int v = 0; v < 8; ++v) {
    bm[2 * v]     = W[(koff + 2 * v) * CH + ncol];
    bm[2 * v + 1] = W[(koff + 2 * v + 1) * CH + ncol];
  }
  return bm;
}

// ---- kNN: one thread per query, LDS-tiled candidates, register top-k ------

template <int K>
__global__ __launch_bounds__(256) void knn_kernel(const float* __restrict__ pos,
                                                  int* __restrict__ idx_out, int n) {
  __shared__ float tile[256][3];
  const int b = blockIdx.y;
  const int i = blockIdx.x * 256 + threadIdx.x;
  const float* pb = pos + (size_t)b * n * 3;
  const bool act = (i < n);
  float qx = 0.f, qy = 0.f, qz = 0.f;
  if (act) { qx = pb[i * 3]; qy = pb[i * 3 + 1]; qz = pb[i * 3 + 2]; }
  float bd[K]; int bi[K];
#pragma unroll
  for (int j = 0; j < K; ++j) { bd[j] = 3.4e38f; bi[j] = 0; }
  for (int t0 = 0; t0 < n; t0 += 256) {
    const int cnt = min(256, n - t0);
    for (int e = threadIdx.x; e < cnt * 3; e += 256)
      tile[e / 3][e % 3] = pb[t0 * 3 + e];
    // warm L2/WGP$ with the next candidate tile while we crunch this one
    if (t0 + 256 < n)
      __builtin_prefetch(&pb[(size_t)(t0 + 256) * 3 + threadIdx.x * 3], 0, 1);
    __syncthreads();
    if (act) {
      for (int c = 0; c < cnt; ++c) {
        float dx = tile[c][0] - qx, dy = tile[c][1] - qy, dz = tile[c][2] - qz;
        float d = dx * dx + dy * dy + dz * dz;
        if (d < bd[K - 1]) {
          int ci = t0 + c;
#pragma unroll
          for (int j = K - 1; j >= 0; --j) {
            if (j > 0 && d < bd[j - 1]) { bd[j] = bd[j - 1]; bi[j] = bi[j - 1]; }
            else { bd[j] = d; bi[j] = ci; break; }
          }
        }
      }
    }
    __syncthreads();
  }
  if (act) {
    int* o = idx_out + ((size_t)b * n + i) * 8;
#pragma unroll
    for (int j = 0; j < K; ++j) o[j] = bi[j];
#pragma unroll
    for (int j = K; j < 8; ++j) o[j] = bi[0];  // pad with self edge (max-neutral)
  }
}

// ---- FPS: one block per graph, LDS-resident, exact argmax tie-break -------

__global__ __launch_bounds__(256) void fps_kernel(const float* __restrict__ pos,
                                                  int* __restrict__ sel, int n, int m) {
  __shared__ float px[2048], py[2048], pz[2048], dd[2048];
  __shared__ float rv[256];
  __shared__ int ri[256];
  const int b = blockIdx.x, t = threadIdx.x;
  const float* pb = pos + (size_t)b * n * 3;
  for (int i = t; i < n; i += 256) { px[i] = pb[i*3]; py[i] = pb[i*3+1]; pz[i] = pb[i*3+2]; }
  __syncthreads();
  const float x0 = px[0], y0 = py[0], z0 = pz[0];
  for (int i = t; i < n; i += 256) {
    float dx = px[i]-x0, dy = py[i]-y0, dz = pz[i]-z0;
    dd[i] = dx*dx + dy*dy + dz*dz;
  }
  if (t == 0) sel[(size_t)b * m] = 0;
  __syncthreads();
  for (int s = 1; s < m; ++s) {
    float bv = -1.f; int bidx = n;
    for (int i = t; i < n; i += 256)
      if (dd[i] > bv) { bv = dd[i]; bidx = i; }
    rv[t] = bv; ri[t] = bidx;
    __syncthreads();
    for (int stride = 128; stride > 0; stride >>= 1) {
      if (t < stride) {
        float ov = rv[t + stride]; int oi = ri[t + stride];
        if (ov > rv[t] || (ov == rv[t] && oi < ri[t])) { rv[t] = ov; ri[t] = oi; }
      }
      __syncthreads();
    }
    const int w = ri[0];
    if (t == 0) sel[(size_t)b * m + s] = w;
    const float wx = px[w], wy = py[w], wz = pz[w];
    for (int i = t; i < n; i += 256) {
      float dx = px[i]-wx, dy = py[i]-wy, dz = pz[i]-wz;
      float nd = dx*dx + dy*dy + dz*dz;
      if (nd < dd[i]) dd[i] = nd;
    }
    __syncthreads();
  }
}

// ---- gather pos+feat by FPS selection -------------------------------------

__global__ __launch_bounds__(256) void gather_kernel(
    const float* __restrict__ p_in, const float* __restrict__ h_in,
    const int* __restrict__ sel, float* __restrict__ p_out,
    float* __restrict__ h_out, int n_in, int m) {
  int tid = blockIdx.x * 256 + threadIdx.x;
  if (tid >= NB * m) return;
  int b = tid / m, i = tid % m;
  int s = sel[(size_t)b * m + i];
#pragma unroll
  for (int c = 0; c < 3; ++c)
    p_out[((size_t)b * m + i) * 3 + c] = p_in[((size_t)b * n_in + s) * 3 + c];
  for (int c = 0; c < CH; ++c)
    h_out[((size_t)b * m + i) * CH + c] = h_in[((size_t)b * n_in + s) * CH + c];
}

// ---- PointNet conv layer: WMMA edge-MLP + in-register max aggregation -----
// 128 threads = 4 waves; each wave owns 2 points x 8 padded edges = 16 WMMA rows.
// W2 (64x64 fp32, contiguous 16KB) is staged Global->LDS by the Tensor Data
// Mover (TENSOR_LOAD_TO_LDS + s_wait_tensorcnt), then converted to bf16.

template <int CH_IN, int KTILES>
__global__ __launch_bounds__(128) void pointnet_kernel(
    const float* __restrict__ h_in, const float* __restrict__ pos,
    const int* __restrict__ idx, const float* __restrict__ W1,
    const float* __restrict__ b1, const float* __restrict__ W2,
    const float* __restrict__ b2, float* __restrict__ h_out, int n) {
  constexpr int KPAD = KTILES * 32;
  constexpr int FIN = 2 * CH_IN + 3;
  __shared__ __bf16 W1s[KPAD][CH];
  __shared__ __bf16 W2s[CH][CH];
  __shared__ float w2stage[CH * CH];
  __shared__ float b1s[CH], b2s[CH];
  __shared__ __bf16 msg[4][16][KPAD];
  __shared__ __bf16 msg2[4][16][CH];

  const int t = threadIdx.x;
  const int wave = t >> 5, lane = t & 31;
  const int hi = lane >> 4, mrow = lane & 15;
  const int bpg = n >> 3;
  const int b = blockIdx.x / bpg;
  const int pbase = (blockIdx.x % bpg) << 3;

#if USE_TDM
  // --- TDM: DMA W2 tile (64x64 f32) into LDS staging; wave 0 issues -------
  if (t < 32) {
    const unsigned long long ga = (unsigned long long)(size_t)W2;
    const unsigned int lds_off = (unsigned int)(size_t)&w2stage[0];  // addr[31:0] = LDS offset
    // D# group0: count=1 | lds_addr | global_addr[56:0] | type=2
    u32x4 g0 = {1u, lds_off, (unsigned int)ga,
                0x80000000u | (unsigned int)((ga >> 32) & 0x01FFFFFFull)};
    // D# group1: data_size=4B; tensor_dim0=64, tensor_dim1=64;
    // tile_dim0=64, tile_dim1=64; tensor_dim0_stride=64; dim1_stride=4096
    i32x8 g1 = {0x00020000, 0x00400000, 0x00400000, 0x00400000,
                0x00000040, 64, 0x10000000, 0};
    i32x4 gz = {0, 0, 0, 0};
#if defined(__clang_major__) && __clang_major__ >= 23
    i32x8 gz8 = {0, 0, 0, 0, 0, 0, 0, 0};
    __builtin_amdgcn_tensor_load_to_lds(g0, g1, gz, gz, gz8, 0);
#else
    __builtin_amdgcn_tensor_load_to_lds(g0, g1, gz, gz, 0);
#endif
    __builtin_amdgcn_s_wait_tensorcnt(0);
  }
#endif

  // stage W1 (bf16, zero-padded rows) + biases into LDS
  for (int i = t; i < KPAD * CH; i += 128) {
    int k = i >> 6, c = i & 63;
    W1s[k][c] = (k < FIN) ? (__bf16)W1[k * CH + c] : (__bf16)0.0f;
  }
#if !USE_TDM
  for (int i = t; i < CH * CH; i += 128) w2stage[i] = W2[i];
#endif
  if (t < CH) { b1s[t] = b1[t]; b2s[t] = b2[t]; }
  __syncthreads();

  // convert TDM-staged W2 to bf16 (visible after syncthreads)
  for (int i = t; i < CH * CH; i += 128) W2s[i >> 6][i & 63] = (__bf16)w2stage[i];

  // build message rows: cat(h_i, h_j, pos_j - pos_i), zero-padded to KPAD
  {
    const int e = mrow;
    const int point = pbase + wave * 2 + (e >> 3);
    const int slot = e & 7;
    const int j = idx[((size_t)b * n + point) * 8 + slot];
    const float* hp = h_in + ((size_t)b * n + point) * CH_IN;
    const float* hj = h_in + ((size_t)b * n + j) * CH_IN;
    const float* pp = pos + ((size_t)b * n + point) * 3;
    const float* pj = pos + ((size_t)b * n + j) * 3;
    float rel[3] = {pj[0] - pp[0], pj[1] - pp[1], pj[2] - pp[2]};
    const int c0 = hi * (KPAD / 2), c1 = c0 + KPAD / 2;
    for (int c = c0; c < c1; ++c) {
      float v;
      if (c < CH_IN) v = hp[c];
      else if (c < 2 * CH_IN) v = hj[c - CH_IN];
      else if (c < FIN) v = rel[c - 2 * CH_IN];
      else v = 0.0f;
      msg[wave][e][c] = (__bf16)v;
    }
  }
  __syncthreads();

  // GEMM1: [16 x KPAD] x [KPAD x 64]
  v8f acc[4] = {};
#pragma unroll
  for (int kt = 0; kt < KTILES; ++kt) {
    v16bf a = load_a_frag(&msg[wave][mrow][0], kt, hi);
#pragma unroll
    for (int nt = 0; nt < 4; ++nt) {
      v16bf bm = load_b_frag(&W1s[0][0], kt, nt, lane);
      acc[nt] = __builtin_amdgcn_wmma_f32_16x16x32_bf16(
          false, a, false, bm, (short)0, acc[nt], false, false);
    }
  }

  // bias + ReLU, respill to LDS as bf16 A-operand for GEMM2
#pragma unroll
  for (int nt = 0; nt < 4; ++nt) {
    float bias = b1s[nt * 16 + mrow];
#pragma unroll
    for (int r = 0; r < 8; ++r) {
      float v = acc[nt][r] + bias;
      v = v > 0.0f ? v : 0.0f;
      msg2[wave][r + 8 * hi][nt * 16 + mrow] = (__bf16)v;
    }
  }
  __syncthreads();

  // GEMM2: [16 x 64] x [64 x 64]
  v8f dacc[4] = {};
#pragma unroll
  for (int kt = 0; kt < 2; ++kt) {
    v16bf a = load_a_frag(&msg2[wave][mrow][0], kt, hi);
#pragma unroll
    for (int nt = 0; nt < 4; ++nt) {
      v16bf bm = load_b_frag(&W2s[0][0], kt, nt, lane);
      dacc[nt] = __builtin_amdgcn_wmma_f32_16x16x32_bf16(
          false, a, false, bm, (short)0, dacc[nt], false, false);
    }
  }

  // rows 0-7 (hi=0 lanes) are point0's 8 edges, rows 8-15 (hi=1) point1's:
  // neighbor max == per-lane max over the 8 accumulator registers.
  const int point = pbase + wave * 2 + hi;
#pragma unroll
  for (int nt = 0; nt < 4; ++nt) {
    float bias = b2s[nt * 16 + mrow];
    float mx = dacc[nt][0] + bias;
#pragma unroll
    for (int r = 1; r < 8; ++r) mx = fmaxf(mx, dacc[nt][r] + bias);
    mx = fmaxf(mx, 0.0f);
    h_out[((size_t)b * n + point) * CH + nt * 16 + mrow] = mx;
  }
}

// ---- global max pool + regression head ------------------------------------

__global__ __launch_bounds__(64) void pool_kernel(const float* __restrict__ h,
                                                  float* __restrict__ pooled, int n) {
  int b = blockIdx.x, c = threadIdx.x;
  float mx = -3.4028235e38f;
  for (int i = 0; i < n; ++i) mx = fmaxf(mx, h[((size_t)b * n + i) * CH + c]);
  pooled[b * CH + c] = mx;
}

__global__ __launch_bounds__(192) void head_kernel(const float* __restrict__ pooled,
                                                   const float* __restrict__ Wr,
                                                   const float* __restrict__ br,
                                                   float* __restrict__ out) {
  int t = threadIdx.x;
  if (t >= NB * 6) return;
  int b = t / 6, o = t % 6;
  float s = br[o];
  for (int c = 0; c < CH; ++c) s += pooled[b * CH + c] * Wr[c * 6 + o];
  out[t] = s;
}

// ---- driver ---------------------------------------------------------------

extern "C" void kernel_launch(void* const* d_in, const int* in_sizes, int n_in,
                              void* d_out, int out_size, void* d_ws, size_t ws_size,
                              hipStream_t stream) {
  const float* pos = (const float*)d_in[1];
  const float* W1a = (const float*)d_in[3];
  const float* b1a = (const float*)d_in[4];
  const float* W1b = (const float*)d_in[5];
  const float* b1b = (const float*)d_in[6];
  const float* W2a = (const float*)d_in[7];
  const float* b2a = (const float*)d_in[8];
  const float* W2b = (const float*)d_in[9];
  const float* b2b = (const float*)d_in[10];
  const float* W3a = (const float*)d_in[11];
  const float* b3a = (const float*)d_in[12];
  const float* W3b = (const float*)d_in[13];
  const float* b3b = (const float*)d_in[14];
  const float* Wr  = (const float*)d_in[15];
  const float* br  = (const float*)d_in[16];

  char* ws = (char*)d_ws;
  size_t off = 0;
  auto alloc = [&](size_t bytes) -> void* {
    void* p = ws + off;
    off = (off + bytes + 255) & ~(size_t)255;
    return p;
  };
  int*   idx1   = (int*)alloc((size_t)NB * 2048 * 8 * 4);
  float* h1     = (float*)alloc((size_t)NB * 2048 * CH * 4);
  int*   sel1   = (int*)alloc((size_t)NB * 1024 * 4);
  float* p2     = (float*)alloc((size_t)NB * 1024 * 3 * 4);
  float* h1g    = (float*)alloc((size_t)NB * 1024 * CH * 4);
  int*   idx2   = (int*)alloc((size_t)NB * 1024 * 8 * 4);
  float* h2     = (float*)alloc((size_t)NB * 1024 * CH * 4);
  int*   sel2   = (int*)alloc((size_t)NB * 512 * 4);
  float* p3     = (float*)alloc((size_t)NB * 512 * 3 * 4);
  float* h2g    = (float*)alloc((size_t)NB * 512 * CH * 4);
  int*   idx3   = (int*)alloc((size_t)NB * 512 * 8 * 4);
  float* h3     = (float*)alloc((size_t)NB * 512 * CH * 4);
  float* pooled = (float*)alloc((size_t)NB * CH * 4);

  // layer 1 (h = pos, k = 6) on 2048 pts
  knn_kernel<6><<<dim3(2048 / 256, NB), 256, 0, stream>>>(pos, idx1, 2048);
  pointnet_kernel<3, 1><<<NB * 2048 / 8, 128, 0, stream>>>(
      pos, pos, idx1, W1a, b1a, W1b, b1b, h1, 2048);

  // FPS 2048 -> 1024, gather
  fps_kernel<<<NB, 256, 0, stream>>>(pos, sel1, 2048, 1024);
  gather_kernel<<<(NB * 1024 + 255) / 256, 256, 0, stream>>>(pos, h1, sel1, p2, h1g, 2048, 1024);

  // layer 2 (k = 4) on 1024 pts
  knn_kernel<4><<<dim3(1024 / 256, NB), 256, 0, stream>>>(p2, idx2, 1024);
  pointnet_kernel<CH, 5><<<NB * 1024 / 8, 128, 0, stream>>>(
      h1g, p2, idx2, W2a, b2a, W2b, b2b, h2, 1024);

  // FPS 1024 -> 512, gather
  fps_kernel<<<NB, 256, 0, stream>>>(p2, sel2, 1024, 512);
  gather_kernel<<<(NB * 512 + 255) / 256, 256, 0, stream>>>(p2, h2, sel2, p3, h2g, 1024, 512);

  // layer 3 (k = 3) on 512 pts
  knn_kernel<3><<<dim3(512 / 256, NB), 256, 0, stream>>>(p3, idx3, 512);
  pointnet_kernel<CH, 5><<<NB * 512 / 8, 128, 0, stream>>>(
      h2g, p3, idx3, W3a, b3a, W3b, b3b, h3, 512);

  // global max pool + head
  pool_kernel<<<NB, 64, 0, stream>>>(h3, pooled, 512);
  head_kernel<<<1, 192, 0, stream>>>(pooled, Wr, br, (float*)d_out);
}
